// DualVSSEncoder_69020124447329
// MI455X (gfx1250) — matrix operand order repaired and checked by reading, not compiled, not run on hardware
//
#include <hip/hip_runtime.h>

// ---------------- problem constants (from reference) ----------------
#define B_SZ     2
#define IN_CH    64
#define OUT_CH   128
#define DI       256        // D_INNER
#define NST      8          // D_STATE
#define RNK      8          // DT_RANK
#define KD       2          // K_DIRS
#define HP       64         // pooled H
#define WP       64         // pooled W
#define L_SEQ    (HP*WP)    // 4096
#define M_ROWS   (B_SZ*L_SEQ) // 8192
#define NPROJ    (KD*(RNK+2*NST)) // 48 live columns
#define NPROJ_PAD 64        // padded to a full BN so the GEMM has no edge cases

#define BM 64               // block tile M
#define BN 64               // block tile N
#define KC 32               // K chunk (one WMMA depth)

typedef __attribute__((ext_vector_type(16))) _Float16 v16h;
typedef __attribute__((ext_vector_type(8)))  float    v8f;

// LDS byte offset of a generic pointer to __shared__ (ISA 10.2: flat->LDS uses addr[31:0]).
__device__ __forceinline__ unsigned lds_off(const void* p) {
  return (unsigned)(unsigned long long)p;
}
// 16B async DMA global -> LDS, tracked by ASYNCcnt (CDNA5 §10 async ops).
__device__ __forceinline__ void async_copy16(unsigned lds_dst, const void* gsrc) {
  asm volatile("global_load_async_to_lds_b128 %0, %1, off"
               :: "v"(lds_dst), "v"(gsrc) : "memory");
}

// ---------------- WMMA f16 GEMM: C[M,N] = A[M,K] * Bt[N,K]^T ----------------
// Block = 256 thr = 8 waves, 64x64 output tile. Double-buffered async LDS staging:
//   stage(c+1) DMA overlaps compute(c); s_wait_asynccnt 2 keeps one stage in flight.
// Wave w: M-subtile (w&3)*16, N-subtiles (w>>2)*32 + {0,16} -> one A frag, two B frags,
// two v_wmma_f32_16x16x32_f16 per chunk.
// Fragment layouts per ISA 7.12.2:
//   A 16x32: lane m(0..15)=row m, halfwords = k {0..7}+8h then {16..23}+8h (h=lane>=16)
//   B 32x16: lane n(0..15)=col n, halfwords = k {0..7}+8h then {16..23}+8h (h=lane>=16)
//   C/D:     float r -> row r+8h, col lane&15
__global__ void __launch_bounds__(256)
vss_wmma_gemm_f16(const _Float16* __restrict__ A,
                  const _Float16* __restrict__ Bt,
                  float* __restrict__ C,
                  int M, int N, int K) {
  __shared__ __align__(16) _Float16 As[2][BM * KC];
  __shared__ __align__(16) _Float16 Bs[2][BN * KC];

  const int t    = threadIdx.x;
  const int nb   = N / BN;
  const int bm   = (blockIdx.x / nb) * BM;
  const int bn   = (blockIdx.x % nb) * BN;

  const int lane = t & 31;
  const int w    = t >> 5;              // wave 0..7
  const int half = lane >> 4;
  const int l15  = lane & 15;
  const int mw   = (w & 3) * 16;        // wave's M offset in block tile
  const int nh   = (w >> 2) * 32;       // wave's N pair offset in block tile

  // staging map: thread t copies halfwords [srow*KC+skk, +8) of each 64x32 stage
  const int srow = t >> 2;              // 0..63
  const int skk  = (t & 3) * 8;         // 0,8,16,24

  const _Float16* gA = A  + (size_t)(bm + srow) * K + skk;
  const _Float16* gB = Bt + (size_t)(bn + srow) * K + skk;
  const unsigned ldsA0 = lds_off(&As[0][srow * KC + skk]);
  const unsigned ldsB0 = lds_off(&Bs[0][srow * KC + skk]);
  const unsigned ldsA1 = lds_off(&As[1][srow * KC + skk]);
  const unsigned ldsB1 = lds_off(&Bs[1][srow * KC + skk]);

  const int nchunk = K / KC;
  // prologue: stage chunk 0 into buffer 0
  async_copy16(ldsA0, gA);
  async_copy16(ldsB0, gB);

  v8f acc0 = {}, acc1 = {};
  for (int c = 0; c < nchunk; ++c) {
    const int cur = c & 1;
    if (c + 1 < nchunk) {               // stage next chunk into the other buffer
      const int k1 = (c + 1) * KC;
      async_copy16(cur ? ldsA0 : ldsA1, gA + k1);
      async_copy16(cur ? ldsB0 : ldsB1, gB + k1);
      asm volatile("s_wait_asynccnt 0x2" ::: "memory");   // chunk c landed
    } else {
      asm volatile("s_wait_asynccnt 0x0" ::: "memory");
    }
    __syncthreads();

    v16h a, b0, b1;
    const _Float16* as  = &As[cur][(mw + l15) * KC];
    const _Float16* bs0 = &Bs[cur][(nh + l15) * KC];
    const _Float16* bs1 = &Bs[cur][(nh + 16 + l15) * KC];
#pragma unroll
    for (int i = 0; i < 8; ++i) a[i]     = as[half * 8 + i];        // 16B contiguous
#pragma unroll
    for (int i = 0; i < 8; ++i) a[8 + i] = as[16 + half * 8 + i];   // 16B contiguous
#pragma unroll
    for (int i = 0; i < 16; ++i) b0[i] = bs0[(i >> 3) * 16 + half * 8 + (i & 7)];
#pragma unroll
    for (int i = 0; i < 16; ++i) b1[i] = bs1[(i >> 3) * 16 + half * 8 + (i & 7)];

    acc0 = __builtin_amdgcn_wmma_f32_16x16x32_f16(false, a, false, b0,
                                                  (short)0, acc0, false, false);
    acc1 = __builtin_amdgcn_wmma_f32_16x16x32_f16(false, a, false, b1,
                                                  (short)0, acc1, false, false);
    __syncthreads();                    // protect buffer reuse at stage c+2
  }

  const int gm = bm + mw + half * 8;
  const int gn = bn + nh + l15;
#pragma unroll
  for (int r = 0; r < 8; ++r) {
    C[(size_t)(gm + r) * N + gn]      = acc0[r];
    C[(size_t)(gm + r) * N + gn + 16] = acc1[r];
  }
}

// ---------------- weight prep: convert to f16, transpose to [N][K], pad x-proj ----------------
__global__ void vss_prep_weights(const float* __restrict__ W_in,   // [128,512]
                                 const float* __restrict__ conv_w, // [128,64,1,1]
                                 const float* __restrict__ W_xproj,// [2,256,24]
                                 const float* __restrict__ W_out,  // [256,128]
                                 _Float16* __restrict__ Wt_in,     // [512,128]
                                 _Float16* __restrict__ Wc,        // [128,64]
                                 _Float16* __restrict__ Wxt,       // [64,256] (rows 48..63 = 0)
                                 _Float16* __restrict__ Wot) {     // [128,256]
  int i = blockIdx.x * blockDim.x + threadIdx.x;
  if (i < 512 * 128) {
    int n = i / 128, k = i % 128;
    Wt_in[i] = (_Float16)W_in[k * 512 + n];
    return;
  }
  i -= 512 * 128;
  if (i < 128 * 64) {                 // conv_w is already [out][in] = [N][K]
    Wc[i] = (_Float16)conv_w[i];
    return;
  }
  i -= 128 * 64;
  if (i < NPROJ_PAD * DI) {
    int n = i / DI, k = i % DI;
    float v = 0.f;
    if (n < NPROJ) {
      int kd = n / 24, j = n % 24;
      v = W_xproj[((size_t)kd * DI + k) * 24 + j];
    }
    Wxt[i] = (_Float16)v;
    return;
  }
  i -= NPROJ_PAD * DI;
  if (i < 128 * DI) {
    int n = i / DI, k = i % DI;
    Wot[i] = (_Float16)W_out[(size_t)k * 128 + n];
  }
}

// ---------------- maxpool 2x2 + f16 convert -> Apool[B*L, 64] ----------------
__global__ void vss_pool(const float* __restrict__ x, _Float16* __restrict__ Apool) {
  int idx = blockIdx.x * blockDim.x + threadIdx.x;
  if (idx >= M_ROWS * IN_CH) return;
  int c  = idx % IN_CH;
  int rl = idx / IN_CH;
  int b = rl / L_SEQ, l = rl % L_SEQ;
  int h = l >> 6, w = l & 63;
  const float* p = x + (((size_t)b * IN_CH + c) * 128 + 2 * h) * 128 + 2 * w;
  float m = fmaxf(fmaxf(p[0], p[1]), fmaxf(p[128], p[129]));
  Apool[(size_t)rl * IN_CH + c] = (_Float16)m;
}

// ---------------- BN(eval)+ReLU -> xin (f32), then LayerNorm -> h (f16) ----------------
__global__ void vss_bn_ln(const float* __restrict__ conv_out,
                          const float* __restrict__ bg, const float* __restrict__ bb,
                          const float* __restrict__ bm, const float* __restrict__ bv,
                          const float* __restrict__ lg, const float* __restrict__ lb,
                          float* __restrict__ xin, _Float16* __restrict__ hbuf) {
  const int row = blockIdx.x;      // 0..M_ROWS-1
  const int c   = threadIdx.x;     // 0..127
  __shared__ float red[OUT_CH];
  float v = conv_out[(size_t)row * OUT_CH + c];
  float scale = bg[c] * rsqrtf(bv[c] + 1e-5f);
  float shift = bb[c] - bm[c] * scale;
  float xv = fmaxf(v * scale + shift, 0.f);
  xin[(size_t)row * OUT_CH + c] = xv;

  red[c] = xv; __syncthreads();
  for (int s = OUT_CH / 2; s > 0; s >>= 1) { if (c < s) red[c] += red[c + s]; __syncthreads(); }
  float mean = red[0] * (1.f / OUT_CH);
  __syncthreads();
  float dv = xv - mean;
  red[c] = dv * dv; __syncthreads();
  for (int s = OUT_CH / 2; s > 0; s >>= 1) { if (c < s) red[c] += red[c + s]; __syncthreads(); }
  float var = red[0] * (1.f / OUT_CH);
  float hn = dv * rsqrtf(var + 1e-5f) * lg[c] + lb[c];
  hbuf[(size_t)row * OUT_CH + c] = (_Float16)hn;
}

// ---------------- depthwise 3x3 SAME + SiLU over x_part = xz[:, :256] ----------------
__global__ void vss_dwconv(const float* __restrict__ xz, const float* __restrict__ dww,
                           float* __restrict__ xc, _Float16* __restrict__ xch) {
  int idx = blockIdx.x * blockDim.x + threadIdx.x;
  if (idx >= M_ROWS * DI) return;
  int d  = idx & (DI - 1);
  int rl = idx >> 8;
  int b = rl / L_SEQ, l = rl % L_SEQ;
  int h = l >> 6, w = l & 63;
  float acc = 0.f;
#pragma unroll
  for (int dy = -1; dy <= 1; ++dy) {
    int hy = h + dy;
    if ((unsigned)hy >= HP) continue;
#pragma unroll
    for (int dx = -1; dx <= 1; ++dx) {
      int wx = w + dx;
      if ((unsigned)wx >= WP) continue;
      acc += xz[((size_t)b * L_SEQ + hy * WP + wx) * (2 * DI) + d] *
             dww[d * 9 + (dy + 1) * 3 + (dx + 1)];
    }
  }
  float s = acc / (1.f + expf(-acc));     // SiLU
  xc[idx]  = s;
  xch[idx] = (_Float16)s;
}

// ---------------- dt = softplus(dt_low @ W_dt + bias), per direction ----------------
__global__ void vss_dt(const float* __restrict__ proj, const float* __restrict__ Wdt,
                       const float* __restrict__ dtb, float* __restrict__ dt) {
  int idx = blockIdx.x * blockDim.x + threadIdx.x;       // k*M_ROWS*DI + rl*DI + d
  if (idx >= KD * M_ROWS * DI) return;
  int d  = idx & (DI - 1);
  int t  = idx >> 8;
  int rl = t % M_ROWS;
  int k  = t / M_ROWS;
  const float* p = proj + (size_t)rl * NPROJ_PAD + k * 24;   // dt_low = first 8 of this dir
  float acc = dtb[k * DI + d];
#pragma unroll
  for (int r = 0; r < RNK; ++r) acc += p[r] * Wdt[(k * RNK + r) * DI + d];
  float sp = (acc > 20.f) ? acc : log1pf(expf(acc));
  dt[idx] = sp;
}

// ---------------- dual selective scan: block = (dir,batch), lane = channel ----------------
// flip(x)@W == flip(x@W), so dir 1 just walks l backwards; y lands at original index.
__global__ void vss_scan(const float* __restrict__ dt, const float* __restrict__ xc,
                         const float* __restrict__ proj, const float* __restrict__ A_log,
                         const float* __restrict__ Dp,
                         float* __restrict__ y0, float* __restrict__ y1) {
  const int k = blockIdx.x >> 1;
  const int b = blockIdx.x & 1;
  const int d = threadIdx.x;           // 0..255
  float An[NST];
#pragma unroll
  for (int n = 0; n < NST; ++n) An[n] = -expf(A_log[((size_t)k * DI + d) * NST + n]);
  const float Dv = Dp[k * DI + d];
  float st[NST];
#pragma unroll
  for (int n = 0; n < NST; ++n) st[n] = 0.f;
  float* __restrict__ yk = (k == 0) ? y0 : y1;

  for (int t = 0; t < L_SEQ; ++t) {
    const int l = (k == 0) ? t : (L_SEQ - 1 - t);
    const size_t row = (size_t)b * L_SEQ + l;
    const float dtv = dt[((size_t)k * M_ROWS + row) * DI + d];
    const float xv  = xc[row * DI + d];
    const float* pr = proj + row * NPROJ_PAD + k * 24;   // [8:16)=Bc, [16:24)=Cc
    const float bx  = dtv * xv;
    float y = xv * Dv;
#pragma unroll
    for (int n = 0; n < NST; ++n) {
      float ab = expf(dtv * An[n]);
      st[n] = ab * st[n] + bx * pr[8 + n];
      y += st[n] * pr[16 + n];
    }
    yk[row * DI + d] = y;
  }
}

// ---------------- out-LayerNorm(y0+y1) * SiLU(z) -> act (f16) ----------------
__global__ void vss_gate(const float* __restrict__ y0, const float* __restrict__ y1,
                         const float* __restrict__ og, const float* __restrict__ ob,
                         const float* __restrict__ xz, _Float16* __restrict__ acth) {
  const int row = blockIdx.x;
  const int d   = threadIdx.x;   // 0..255
  __shared__ float red[DI];
  float ys = y0[(size_t)row * DI + d] + y1[(size_t)row * DI + d];
  red[d] = ys; __syncthreads();
  for (int s = DI / 2; s > 0; s >>= 1) { if (d < s) red[d] += red[d + s]; __syncthreads(); }
  float mean = red[0] * (1.f / DI);
  __syncthreads();
  float dv = ys - mean;
  red[d] = dv * dv; __syncthreads();
  for (int s = DI / 2; s > 0; s >>= 1) { if (d < s) red[d] += red[d + s]; __syncthreads(); }
  float var = red[0] * (1.f / DI);
  float yn = dv * rsqrtf(var + 1e-5f) * og[d] + ob[d];
  float z  = xz[(size_t)row * (2 * DI) + DI + d];
  float g  = z / (1.f + expf(-z));
  acth[(size_t)row * DI + d] = (_Float16)(yn * g);
}

// ---------------- + residual, BHWC -> BCHW, duplicate tuple output ----------------
__global__ void vss_epilogue(const float* __restrict__ outmat, const float* __restrict__ xin,
                             float* __restrict__ out) {
  int idx = blockIdx.x * blockDim.x + threadIdx.x;
  if (idx >= M_ROWS * OUT_CH) return;
  int c  = idx & (OUT_CH - 1);
  int rl = idx >> 7;
  int b = rl / L_SEQ, l = rl % L_SEQ;
  int h = l >> 6, w = l & 63;
  float v = outmat[idx] + xin[idx];
  size_t o = (((size_t)b * OUT_CH + c) * HP + h) * WP + w;
  out[o] = v;
  out[(size_t)B_SZ * OUT_CH * L_SEQ + o] = v;   // second element of returned tuple
}

// ======================= host-side launcher =======================
extern "C" void kernel_launch(void* const* d_in, const int* in_sizes, int n_in,
                              void* d_out, int out_size, void* d_ws, size_t ws_size,
                              hipStream_t stream) {
  (void)in_sizes; (void)n_in; (void)out_size; (void)ws_size;
  const float* x_in    = (const float*)d_in[0];
  const float* conv_w  = (const float*)d_in[1];
  const float* bn_g    = (const float*)d_in[2];
  const float* bn_b    = (const float*)d_in[3];
  const float* bn_m    = (const float*)d_in[4];
  const float* bn_v    = (const float*)d_in[5];
  const float* ln_g    = (const float*)d_in[6];
  const float* ln_b    = (const float*)d_in[7];
  const float* W_in    = (const float*)d_in[8];
  const float* dw_w    = (const float*)d_in[9];
  const float* W_xproj = (const float*)d_in[10];
  const float* W_dt    = (const float*)d_in[11];
  const float* dt_bias = (const float*)d_in[12];
  const float* A_log   = (const float*)d_in[13];
  const float* Dp      = (const float*)d_in[14];
  const float* on_g    = (const float*)d_in[15];
  const float* on_b    = (const float*)d_in[16];
  const float* W_out   = (const float*)d_in[17];
  float* out = (float*)d_out;

  // scratch carve-out (256B aligned)
  char* ws = (char*)d_ws;
  size_t off = 0;
  auto alloc = [&](size_t bytes) -> void* {
    void* p = ws + off;
    off = (off + bytes + 255) & ~(size_t)255;
    return p;
  };
  _Float16* Apool   = (_Float16*)alloc((size_t)M_ROWS * IN_CH * 2);
  _Float16* Wc      = (_Float16*)alloc((size_t)OUT_CH * IN_CH * 2);
  _Float16* Wt_in   = (_Float16*)alloc((size_t)512 * 128 * 2);
  _Float16* Wxt     = (_Float16*)alloc((size_t)NPROJ_PAD * DI * 2);
  _Float16* Wot     = (_Float16*)alloc((size_t)128 * DI * 2);
  float*    conv_o  = (float*)   alloc((size_t)M_ROWS * OUT_CH * 4);
  float*    xin     = (float*)   alloc((size_t)M_ROWS * OUT_CH * 4);
  _Float16* hbuf    = (_Float16*)alloc((size_t)M_ROWS * OUT_CH * 2);
  float*    xz      = (float*)   alloc((size_t)M_ROWS * 2 * DI * 4);
  float*    xc      = (float*)   alloc((size_t)M_ROWS * DI * 4);
  _Float16* xch     = (_Float16*)alloc((size_t)M_ROWS * DI * 2);
  float*    proj    = (float*)   alloc((size_t)M_ROWS * NPROJ_PAD * 4);
  float*    dtb     = (float*)   alloc((size_t)KD * M_ROWS * DI * 4);
  float*    y0      = (float*)   alloc((size_t)M_ROWS * DI * 4);
  float*    y1      = (float*)   alloc((size_t)M_ROWS * DI * 4);
  _Float16* acth    = (_Float16*)alloc((size_t)M_ROWS * DI * 2);
  float*    outmat  = (float*)   alloc((size_t)M_ROWS * OUT_CH * 4);

  auto gemm = [&](const _Float16* A, const _Float16* Bt, float* C, int M, int N, int K) {
    int blocks = (M / BM) * (N / BN);
    vss_wmma_gemm_f16<<<blocks, 256, 0, stream>>>(A, Bt, C, M, N, K);
  };

  // 1) weight conversion (f16, transposed, x-proj padded to 64 columns)
  {
    int tot = 512 * 128 + 128 * 64 + NPROJ_PAD * DI + 128 * DI;
    vss_prep_weights<<<(tot + 255) / 256, 256, 0, stream>>>(
        W_in, conv_w, W_xproj, W_out, Wt_in, Wc, Wxt, Wot);
  }
  // 2) maxpool -> f16 activation matrix
  vss_pool<<<(M_ROWS * IN_CH + 255) / 256, 256, 0, stream>>>(x_in, Apool);
  // 3) 1x1 conv GEMM (WMMA + async-LDS double buffering)
  gemm(Apool, Wc, conv_o, M_ROWS, OUT_CH, IN_CH);
  // 4) BN + ReLU + LayerNorm
  vss_bn_ln<<<M_ROWS, OUT_CH, 0, stream>>>(conv_o, bn_g, bn_b, bn_m, bn_v, ln_g, ln_b, xin, hbuf);
  // 5) in-projection GEMM (WMMA): [8192,128] x [128,512]
  gemm(hbuf, Wt_in, xz, M_ROWS, 2 * DI, OUT_CH);
  // 6) depthwise 3x3 + SiLU
  vss_dwconv<<<(M_ROWS * DI + 255) / 256, 256, 0, stream>>>(xz, dw_w, xc, xch);
  // 7) fused dual-direction x-projection GEMM (WMMA): [8192,256] x [256,64(pad)]
  gemm(xch, Wxt, proj, M_ROWS, NPROJ_PAD, DI);
  // 8) dt = softplus(low-rank proj)
  vss_dt<<<(KD * M_ROWS * DI + 255) / 256, 256, 0, stream>>>(proj, W_dt, dt_bias, dtb);
  // 9) dual selective scan (fwd + bwd), states in registers
  vss_scan<<<KD * B_SZ, DI, 0, stream>>>(dtb, xc, proj, A_log, Dp, y0, y1);
  // 10) out-LN * SiLU(z) gate -> f16
  vss_gate<<<M_ROWS, DI, 0, stream>>>(y0, y1, on_g, on_b, xz, acth);
  // 11) out-projection GEMM (WMMA): [8192,256] x [256,128]
  gemm(acth, Wot, outmat, M_ROWS, OUT_CH, DI);
  // 12) residual + layout + tuple duplication
  vss_epilogue<<<(M_ROWS * OUT_CH + 255) / 256, 256, 0, stream>>>(outmat, xin, out);
}
